// BioEncoderMamba_80676665688199
// MI455X (gfx1250) — matrix operand order repaired and checked by reading, not compiled, not run on hardware
//
#include <hip/hip_runtime.h>
#include <hip/hip_bf16.h>
#include <math.h>

// ---------------------------------------------------------------------------
// BioEncoderMamba on gfx1250: bf16 WMMA GEMMs with TDM(LDS)-staged B panels,
// plus VALU scan/conv/LN kernels.
// Shapes: B=256, C_IN=7, T=160, PATCH=10, L=16, D_MODEL=512, D_INNER=1024,
//         DT_RANK=32, D_STATE=16, D_CONV=4, N_LAYERS=2, M = B*L = 4096.
// ---------------------------------------------------------------------------

typedef __attribute__((ext_vector_type(16))) __bf16 v16bf;
typedef __attribute__((ext_vector_type(8)))  float  v8f;
typedef __attribute__((ext_vector_type(4)))  unsigned int v4u;
typedef __attribute__((ext_vector_type(8)))  int    v8i;
typedef __attribute__((ext_vector_type(4)))  int    v4i;

#define EPSC 1e-5f
#define ACT_NONE     0
#define ACT_SOFTPLUS 1
#define ACT_ADD      2

#define MROWS 4096   // B * L_TOK

// ---------------------------------------------------------------------------
// Issue a 1-D TDM copy of `ndw` dwords from global `gptr` to LDS offset
// `lds_addr` (wave-relative bytes).  D# layout per CDNA5 ISA ch.8:
//   group0: count=1 | lds_addr | global_addr[56:0] | type=2
//   group1: data_size=4B, tensor_dim0=tile_dim0=ndw, dim1 unused
// Tracked by TENSORcnt.  6-arg builtin form (clang-23 / therock-10.0).
// ---------------------------------------------------------------------------
__device__ __forceinline__ void tdm_load_1d(unsigned int lds_addr,
                                            const void* gptr, unsigned int ndw)
{
    const unsigned long long ga = (unsigned long long)(uintptr_t)gptr;
    v4u g0;
    g0[0] = 1u;                                   // count=1, user desc
    g0[1] = lds_addr;                             // LDS byte address
    g0[2] = (unsigned int)ga;                     // global_addr[31:0]
    g0[3] = (unsigned int)((ga >> 32) & 0x01FFFFFFu) | (2u << 30); // [56:32]|type=2
    v8i g1;
    g1[0] = (int)(2u << 16);                      // data_size = 4 bytes
    g1[1] = (int)((ndw & 0xFFFFu) << 16);         // tensor_dim0[15:0]
    g1[2] = (int)(((ndw >> 16) & 0xFFFFu) | (1u << 16)); // tdim0[31:16], tdim1=1
    g1[3] = (int)((ndw & 0xFFFFu) << 16);         // tile_dim0 = ndw
    g1[4] = 0;                                    // tile_dim1/2 unused
    g1[5] = (int)ndw;                             // tensor_dim0_stride[31:0]
    g1[6] = 0;
    g1[7] = 0;
    const v4i z4 = {0, 0, 0, 0};
    const v8i z8 = {0, 0, 0, 0, 0, 0, 0, 0};
    __builtin_amdgcn_tensor_load_to_lds(g0, g1, z4, z4, z8, 0);
}

// ---------------------------------------------------------------------------
// Pack f32 weight matrix (K x N, or transposed N x K) into bf16 B-fragments
// for V_WMMA_F32_16X16X32_BF16 SRC1 (k order contiguous, same as A).  Fragment order is
// f = kt * (N/16) + nTile  so that one k-slice of consecutive nTiles is a
// CONTIGUOUS panel (TDM-friendly).  Within fragment: lane(0..31) x 16 elems,
// column n = nTile*16 + (lane&15), element e -> K = kt*32 + ((lane>>4)<<4)+e.
// Zero-pads K >= Kreal (patch embed K=70 -> 96).
// ---------------------------------------------------------------------------
__global__ void pack_w_k(const float* __restrict__ W, __hip_bfloat16* __restrict__ P,
                         int Kpad, int N, int Kreal, int transposed)
{
    const int Kt = Kpad >> 5;
    const int nTilesTotal = N >> 4;
    const size_t tid = (size_t)blockIdx.x * blockDim.x + threadIdx.x;
    const size_t total = (size_t)nTilesTotal * (size_t)Kt * 32u;
    if (tid >= total) return;
    const int lane  = (int)(tid & 31);
    const size_t f  = tid >> 5;                   // = kt*nTilesTotal + nTile
    const int nTile = (int)(f % (size_t)nTilesTotal);
    const int kt    = (int)(f / (size_t)nTilesTotal);
    const int half  = lane >> 4;
    const int n     = nTile * 16 + (lane & 15);
    __hip_bfloat16* dst = P + tid * 16;
    for (int e = 0; e < 16; ++e) {
        const int k = kt * 32 + (half << 4) + e;
        float v = 0.f;
        if (k < Kreal)
            v = transposed ? W[(size_t)n * Kreal + k] : W[(size_t)k * N + n];
        dst[e] = __float2bfloat16(v);
    }
}

// ---------------------------------------------------------------------------
// GEMM: C(M x N) = A(f32, M x K) * Bpacked(bf16) with fused epilogue.
// Block = 4 waves; wave w owns mTile = blockIdx.y*4 + w (16 rows) and NT
// consecutive 16-wide column tiles (shared across the block via LDS panel).
// Per k-step: wave0 TDM-loads the NEXT 32xN-panel (NT KB) into LDS while all
// waves consume the CURRENT one (double buffered), then s_wait_tensorcnt +
// barrier.  A fragment: four coalesced float4 loads, f32->bf16 in-register,
// reused across NT back-to-back WMMAs.
// ---------------------------------------------------------------------------
template <int NT>
__global__ __launch_bounds__(128)
void gemm_bf16_wmma_k(const float* __restrict__ A, int lda,
                      const __hip_bfloat16* __restrict__ Bp,
                      const float* __restrict__ bias,
                      float* __restrict__ C, int ldc,
                      int Kt, int nTilesTotal, int act, int revM)
{
    __shared__ __align__(32) unsigned char smem[2][NT * 1024];

    const int lane = threadIdx.x & 31;
    const int wave = threadIdx.x >> 5;
    const int half = lane >> 4;
    const int l15  = lane & 15;
    const int mTile = blockIdx.y * 4 + wave;
    const int nt0   = blockIdx.x * NT;

    const float* Arow = A + (size_t)(mTile * 16 + l15) * lda;

    v8f acc[NT];
#pragma unroll
    for (int nt = 0; nt < NT; ++nt)
        acc[nt] = (v8f){0.f, 0.f, 0.f, 0.f, 0.f, 0.f, 0.f, 0.f};

    // Preload panel for kt=0.
    if (wave == 0) {
        tdm_load_1d((unsigned int)(uintptr_t)&smem[0][0],
                    Bp + ((size_t)0 * nTilesTotal + nt0) * 512, NT * 256u);
        __builtin_amdgcn_s_wait_tensorcnt(0);
    }
    __syncthreads();

    for (int kt = 0; kt < Kt; ++kt) {
        const int cur = kt & 1;
        // Prefetch next panel via TDM while computing on the current one.
        if (wave == 0 && kt + 1 < Kt)
            tdm_load_1d((unsigned int)(uintptr_t)&smem[cur ^ 1][0],
                        Bp + ((size_t)(kt + 1) * nTilesTotal + nt0) * 512, NT * 256u);

        // A fragment: row = lane&15,
        // elements 0..15 -> K = kb..kb+15 (matches packed B),
        // kb = kt*32 + (half<<4).
        const int kb = kt * 32 + (half << 4);
        const float4 a0 = *(const float4*)(Arow + kb + 0);
        const float4 a1 = *(const float4*)(Arow + kb + 4);
        const float4 a2 = *(const float4*)(Arow + kb + 8);
        const float4 a3 = *(const float4*)(Arow + kb + 12);
        v16bf a;
        a[0]  = (__bf16)a0.x; a[1]  = (__bf16)a0.y; a[2]  = (__bf16)a0.z; a[3]  = (__bf16)a0.w;
        a[4]  = (__bf16)a1.x; a[5]  = (__bf16)a1.y; a[6]  = (__bf16)a1.z; a[7]  = (__bf16)a1.w;
        a[8]  = (__bf16)a2.x; a[9]  = (__bf16)a2.y; a[10] = (__bf16)a2.z; a[11] = (__bf16)a2.w;
        a[12] = (__bf16)a3.x; a[13] = (__bf16)a3.y; a[14] = (__bf16)a3.z; a[15] = (__bf16)a3.w;

#pragma unroll
        for (int nt = 0; nt < NT; ++nt) {
            const v16bf b = *(const v16bf*)(&smem[cur][((size_t)nt * 32 + lane) * 32]);
            acc[nt] = __builtin_amdgcn_wmma_f32_16x16x32_bf16(
                false, a, false, b, (short)0, acc[nt], false, false);
        }

        if (wave == 0 && kt + 1 < Kt)
            __builtin_amdgcn_s_wait_tensorcnt(0);
        __syncthreads();
    }

    // Epilogue: bias / softplus / residual-accumulate / row time-reversal.
#pragma unroll
    for (int nt = 0; nt < NT; ++nt) {
        const int col = (nt0 + nt) * 16 + l15;
        const float bv = bias ? bias[col] : 0.f;
#pragma unroll
        for (int r = 0; r < 8; ++r) {
            int m = mTile * 16 + (half << 3) + r;       // C/D layout
            if (revM) m = (m & ~15) | (15 - (m & 15));  // fold [:, ::-1]
            float v = acc[nt][r] + bv;
            if (act == ACT_SOFTPLUS) v = (v > 20.f) ? v : log1pf(expf(v));
            float* dst = C + (size_t)m * ldc + col;
            if (act == ACT_ADD) *dst = *dst + v; else *dst = v;
        }
    }
}

// ---------------------------------------------------------------------------
// BatchNorm over channels + patchify into padded (4096 x 96) matrix (K=70 real)
// ---------------------------------------------------------------------------
__global__ void bn_patch_k(const float* __restrict__ x,
                           const float* __restrict__ gamma, const float* __restrict__ beta,
                           const float* __restrict__ mean,  const float* __restrict__ var,
                           float* __restrict__ ppad)
{
    const int idx = blockIdx.x * blockDim.x + threadIdx.x; // m*96 + k
    if (idx >= MROWS * 96) return;
    const int m = idx / 96, k = idx - m * 96;
    float v = 0.f;
    if (k < 70) {
        const int c = k / 10, j = k - c * 10;
        const int bb = m >> 4, t = m & 15;
        const float xv = x[((size_t)bb * 7 + c) * 160 + t * 10 + j];
        v = (xv - mean[c]) * rsqrtf(var[c] + EPSC) * gamma[c] + beta[c];
    }
    ppad[idx] = v;
}

// ---------------------------------------------------------------------------
// LayerNorm over D=512 (one block per row), optional positional add after LN.
// ---------------------------------------------------------------------------
__global__ __launch_bounds__(256)
void ln512_k(const float* __restrict__ in, const float* __restrict__ g,
             const float* __restrict__ b,  const float* __restrict__ pos,
             float* __restrict__ out)
{
    const int m = blockIdx.x;
    const int t = threadIdx.x;
    const float* row = in + (size_t)m * 512;
    const float x0 = row[t], x1 = row[t + 256];
    __shared__ float s1[256], s2[256];
    s1[t] = x0 + x1;
    s2[t] = x0 * x0 + x1 * x1;
    __syncthreads();
    for (int s = 128; s > 0; s >>= 1) {
        if (t < s) { s1[t] += s1[t + s]; s2[t] += s2[t + s]; }
        __syncthreads();
    }
    const float mu   = s1[0] * (1.f / 512.f);
    const float varv = s2[0] * (1.f / 512.f) - mu * mu;
    const float rstd = rsqrtf(varv + EPSC);
    float y0 = (x0 - mu) * rstd * g[t]       + b[t];
    float y1 = (x1 - mu) * rstd * g[t + 256] + b[t + 256];
    if (pos) {
        const int tok = m & 15;
        y0 += pos[tok * 512 + t];
        y1 += pos[tok * 512 + t + 256];
    }
    float* orow = out + (size_t)m * 512;
    orow[t] = y0; orow[t + 256] = y1;
}

// ---------------------------------------------------------------------------
// Reverse token axis (t -> 15-t) of a (4096 x 512) activation.
// ---------------------------------------------------------------------------
__global__ void reverse_k(const float* __restrict__ in, float* __restrict__ out)
{
    const int idx = blockIdx.x * blockDim.x + threadIdx.x; // m*512 + d
    if (idx >= MROWS * 512) return;
    const int m = idx >> 9, d = idx & 511;
    const int mr = (m & ~15) | (15 - (m & 15));
    out[(size_t)m * 512 + d] = in[(size_t)mr * 512 + d];
}

// ---------------------------------------------------------------------------
// Depthwise causal conv (width 4) over u = uz[:, 0:1024] + SiLU -> uc.
// ---------------------------------------------------------------------------
__global__ void conv_silu_k(const float* __restrict__ uz,
                            const float* __restrict__ cw, const float* __restrict__ cb,
                            float* __restrict__ uc)
{
    const int idx = blockIdx.x * blockDim.x + threadIdx.x; // m*1024 + d
    if (idx >= MROWS * 1024) return;
    const int m = idx >> 10, d = idx & 1023;
    const int bb = m >> 4, t = m & 15;
    float acc = cb[d];
#pragma unroll
    for (int k = 0; k < 4; ++k) {
        const int tt = t + k - 3;
        if (tt >= 0) acc += uz[((size_t)(bb * 16 + tt)) * 2048 + d] * cw[d * 4 + k];
    }
    uc[(size_t)m * 1024 + d] = acc / (1.f + expf(-acc)); // silu
}

// ---------------------------------------------------------------------------
// Selective scan, one thread per (batch, channel); 16 steps x 16 states.
// proj rows: [0:32]=dtr (already consumed), [32:48]=B, [48:64]=C.
// Fuses the +uc*D skip and the silu(z) gate so output feeds out_proj directly.
// ---------------------------------------------------------------------------
__global__ void scan_k(const float* __restrict__ uc,   const float* __restrict__ proj,
                       const float* __restrict__ dtb,  const float* __restrict__ uz,
                       const float* __restrict__ A_log, const float* __restrict__ Dskip,
                       float* __restrict__ gated)
{
    const int idx = blockIdx.x * blockDim.x + threadIdx.x; // b*1024 + d
    if (idx >= 256 * 1024) return;
    const int bb = idx >> 10, d = idx & 1023;

    float Ac[16], hst[16];
#pragma unroll
    for (int n = 0; n < 16; ++n) { Ac[n] = -expf(A_log[d * 16 + n]); hst[n] = 0.f; }
    const float Dv = Dskip[d];

    for (int t = 0; t < 16; ++t) {
        const int m = bb * 16 + t;
        const float dtv = dtb[(size_t)m * 1024 + d];
        const float ucv = uc[(size_t)m * 1024 + d];
        const float zv  = uz[(size_t)m * 2048 + 1024 + d];
        const float du  = dtv * ucv;
        const float* pr = proj + (size_t)m * 64;
        float y = 0.f;
#pragma unroll
        for (int n = 0; n < 16; ++n) {
            hst[n] = expf(dtv * Ac[n]) * hst[n] + du * pr[32 + n];
            y += hst[n] * pr[48 + n];
        }
        y += ucv * Dv;
        gated[(size_t)m * 1024 + d] = y * (zv / (1.f + expf(-zv)));
    }
}

// ---------------------------------------------------------------------------
// Host-side orchestration.
// ---------------------------------------------------------------------------
extern "C" void kernel_launch(void* const* d_in, const int* in_sizes, int n_in,
                              void* d_out, int out_size, void* d_ws, size_t ws_size,
                              hipStream_t stream)
{
    (void)in_sizes; (void)n_in; (void)out_size; (void)ws_size;

    const float* x          = (const float*)d_in[0];
    const float* bn_gamma   = (const float*)d_in[1];
    const float* bn_beta    = (const float*)d_in[2];
    const float* bn_mean    = (const float*)d_in[3];
    const float* bn_var     = (const float*)d_in[4];
    const float* patch_w    = (const float*)d_in[5];   // (512, 70)
    const float* patch_b    = (const float*)d_in[6];
    const float* ln_patch_g = (const float*)d_in[7];
    const float* ln_patch_b = (const float*)d_in[8];
    const float* pos        = (const float*)d_in[9];   // (1,16,512)
    const float* blk_ln_g   = (const float*)d_in[10];  // (2,512)
    const float* blk_ln_b   = (const float*)d_in[11];
    const float* in_proj_w  = (const float*)d_in[12];  // (2,2,512,2048)
    const float* conv_w     = (const float*)d_in[13];  // (2,2,1024,4)
    const float* conv_b     = (const float*)d_in[14];  // (2,2,1024)
    const float* x_proj_w   = (const float*)d_in[15];  // (2,2,1024,64)
    const float* dt_proj_w  = (const float*)d_in[16];  // (2,2,32,1024)
    const float* dt_proj_b  = (const float*)d_in[17];  // (2,2,1024)
    const float* A_log      = (const float*)d_in[18];  // (2,2,1024,16)
    const float* Dskip      = (const float*)d_in[19];  // (2,2,1024)
    const float* out_proj_w = (const float*)d_in[20];  // (2,2,1024,512)
    const float* final_ln_g = (const float*)d_in[21];
    const float* final_ln_b = (const float*)d_in[22];
    float* out = (float*)d_out;                        // (256,16,512)

    // ---- workspace carving (256B aligned) ----
    char* ws = (char*)d_ws;
    size_t off = 0;
    auto carve = [&](size_t bytes) -> void* {
        void* p = ws + off;
        off += (bytes + 255) & ~(size_t)255;
        return p;
    };
    float* ppad  = (float*)carve((size_t)MROWS * 96 * 4);
    float* tmp   = (float*)carve((size_t)MROWS * 512 * 4);
    float* h     = (float*)carve((size_t)MROWS * 512 * 4);
    float* hn    = (float*)carve((size_t)MROWS * 512 * 4);
    float* hnrev = (float*)carve((size_t)MROWS * 512 * 4);
    float* uz    = (float*)carve((size_t)MROWS * 2048 * 4);
    float* uc    = (float*)carve((size_t)MROWS * 1024 * 4);
    float* proj  = (float*)carve((size_t)MROWS * 64 * 4);
    float* dtb   = (float*)carve((size_t)MROWS * 1024 * 4);
    float* gated = (float*)carve((size_t)MROWS * 1024 * 4);
    __hip_bfloat16* patchP = (__hip_bfloat16*)carve((size_t)96 * 512 * 2);
    __hip_bfloat16* inP    = (__hip_bfloat16*)carve((size_t)4 * 512 * 2048 * 2);
    __hip_bfloat16* xpP    = (__hip_bfloat16*)carve((size_t)4 * 1024 * 64 * 2);
    __hip_bfloat16* dtP    = (__hip_bfloat16*)carve((size_t)4 * 32 * 1024 * 2);
    __hip_bfloat16* outP   = (__hip_bfloat16*)carve((size_t)4 * 1024 * 512 * 2);

    auto packN = [&](const float* W, __hip_bfloat16* P, int Kpad, int N, int Kreal, int tr) {
        const size_t total = (size_t)(N / 16) * (Kpad / 32) * 32;
        const int blocks = (int)((total + 255) / 256);
        hipLaunchKernelGGL(pack_w_k, dim3(blocks), dim3(256), 0, stream, W, P, Kpad, N, Kreal, tr);
    };

    // ---- pack all weights into WMMA B-fragment layout (bf16, TDM panels) ----
    packN(patch_w, patchP, 96, 512, 70, /*transposed=*/1);
    for (int li = 0; li < 4; ++li) { // li = layer*2 + dir
        packN(in_proj_w  + (size_t)li * 512 * 2048, inP  + (size_t)li * 512 * 2048, 512,  2048, 512,  0);
        packN(x_proj_w   + (size_t)li * 1024 * 64,  xpP  + (size_t)li * 1024 * 64,  1024, 64,   1024, 0);
        packN(dt_proj_w  + (size_t)li * 32 * 1024,  dtP  + (size_t)li * 32 * 1024,  32,   1024, 32,   0);
        packN(out_proj_w + (size_t)li * 1024 * 512, outP + (size_t)li * 1024 * 512, 1024, 512,  1024, 0);
    }

    // NT=8 (128 columns/wave) for wide GEMMs, NT=4 for N=64.
    auto gemm8 = [&](const float* A, int lda, const __hip_bfloat16* Bp, const float* bias,
                     float* C, int ldc, int N, int Kt, int act, int revM) {
        hipLaunchKernelGGL(gemm_bf16_wmma_k<8>, dim3(N / 128, MROWS / 64), dim3(128), 0, stream,
                           A, lda, Bp, bias, C, ldc, Kt, N / 16, act, revM);
    };
    auto gemm4 = [&](const float* A, int lda, const __hip_bfloat16* Bp, const float* bias,
                     float* C, int ldc, int N, int Kt, int act, int revM) {
        hipLaunchKernelGGL(gemm_bf16_wmma_k<4>, dim3(N / 64, MROWS / 64), dim3(128), 0, stream,
                           A, lda, Bp, bias, C, ldc, Kt, N / 16, act, revM);
    };

    // ---- patch embedding: BN + patchify -> GEMM(K=96) -> LN(+pos) -> h ----
    hipLaunchKernelGGL(bn_patch_k, dim3((MROWS * 96 + 255) / 256), dim3(256), 0, stream,
                       x, bn_gamma, bn_beta, bn_mean, bn_var, ppad);
    gemm8(ppad, 96, patchP, patch_b, tmp, 512, 512, 3, ACT_NONE, 0);
    hipLaunchKernelGGL(ln512_k, dim3(MROWS), dim3(256), 0, stream,
                       tmp, ln_patch_g, ln_patch_b, pos, h);

    // ---- Mamba layers ----
    for (int l = 0; l < 2; ++l) {
        hipLaunchKernelGGL(ln512_k, dim3(MROWS), dim3(256), 0, stream,
                           h, blk_ln_g + l * 512, blk_ln_b + l * 512, (const float*)nullptr, hn);
        hipLaunchKernelGGL(reverse_k, dim3((MROWS * 512 + 255) / 256), dim3(256), 0, stream, hn, hnrev);

        for (int dir = 0; dir < 2; ++dir) {
            const int li = l * 2 + dir;
            const float* Ain = dir ? hnrev : hn;

            // in_proj: (4096,512) x (512,2048) -> uz = [u | z]
            gemm8(Ain, 512, inP + (size_t)li * 512 * 2048, nullptr, uz, 2048, 2048, 16, ACT_NONE, 0);

            // depthwise conv + silu -> uc
            hipLaunchKernelGGL(conv_silu_k, dim3((MROWS * 1024 + 255) / 256), dim3(256), 0, stream,
                               uz, conv_w + (size_t)li * 1024 * 4, conv_b + (size_t)li * 1024, uc);

            // x_proj: (4096,1024) x (1024,64) -> proj = [dtr | B | C]
            gemm4(uc, 1024, xpP + (size_t)li * 1024 * 64, nullptr, proj, 64, 64, 32, ACT_NONE, 0);

            // dt_proj + bias + softplus: (4096, 32 of proj) x (32,1024) -> dtb
            gemm8(proj, 64, dtP + (size_t)li * 32 * 1024, dt_proj_b + (size_t)li * 1024,
                  dtb, 1024, 1024, 1, ACT_SOFTPLUS, 0);

            // selective scan + D-skip + silu(z) gate -> gated
            hipLaunchKernelGGL(scan_k, dim3((256 * 1024 + 255) / 256), dim3(256), 0, stream,
                               uc, proj, dtb, uz,
                               A_log + (size_t)li * 1024 * 16, Dskip + (size_t)li * 1024, gated);

            // out_proj: (4096,1024) x (1024,512), accumulate into residual h;
            // backward direction un-reverses rows at store time.
            gemm8(gated, 1024, outP + (size_t)li * 1024 * 512, nullptr, h, 512, 512, 32, ACT_ADD, dir);
        }
    }

    // ---- final LayerNorm -> output ----
    hipLaunchKernelGGL(ln512_k, dim3(MROWS), dim3(256), 0, stream,
                       h, final_ln_g, final_ln_b, (const float*)nullptr, out);
}
